// M_1803886264934
// MI455X (gfx1250) — matrix operand order repaired and checked by reading, not compiled, not run on hardware
//
#include <hip/hip_runtime.h>
#include <math.h>
#include <stdint.h>

// Problem constants (from reference): B*H*W = 256*7*7 = 12544 rows, C=1024, F=4096
#define M_ROWS 12544
#define K_DIM  1024
#define N_DIM  4096

// GEMM tiling: block = 64(M) x 256(N), K-step 32; 8 waves, each owns 32x64
#define BM 64
#define BN 256
#define BK 32
#define LDA 40   // padded LDS row stride (bf16 elems): 80B, 16B-aligned, kills bank conflicts
#define LDB 40
#define KSTEPS (K_DIM / BK)   // 32

typedef __attribute__((ext_vector_type(16))) __bf16 bfrag_t;   // WMMA A/B fragment (16 bf16)
typedef __attribute__((ext_vector_type(8)))  float  v8f;       // WMMA C/D fragment

// ---- CDNA5 async global->LDS path (guarded; falls back to reg-staged copy) --
#if defined(__has_builtin)
#if __has_builtin(__builtin_amdgcn_global_load_async_to_lds_b128) && \
    __has_builtin(__builtin_amdgcn_s_wait_asynccnt)
#define USE_ASYNC_LDS 1
#endif
#endif

#ifdef USE_ASYNC_LDS
typedef int v4i_ __attribute__((ext_vector_type(4)));
typedef __attribute__((address_space(1))) v4i_* as1_v4i_ptr;   // global (device) src
typedef __attribute__((address_space(3))) v4i_* as3_v4i_ptr;   // LDS dst

__device__ __forceinline__ void async_cp16(const void* g, void* l) {
  // generic->AS(1)/AS(3) via integer casts (low 32 bits of a generic shared
  // pointer are the LDS offset)
  __builtin_amdgcn_global_load_async_to_lds_b128(
      (as1_v4i_ptr)(uintptr_t)g,
      (as3_v4i_ptr)(unsigned int)(uintptr_t)l,
      0, 0);
}
#endif

__device__ __forceinline__ unsigned short f32_to_bf16(float f) {
  unsigned int u = __float_as_uint(f);
  unsigned int r = 0x7FFFu + ((u >> 16) & 1u);   // round-to-nearest-even
  return (unsigned short)((u + r) >> 16);
}

__device__ __forceinline__ float gelu_erf(float y) {
  return 0.5f * y * (1.0f + erff(y * 0.70710678118654752f));
}

// ---------------------------------------------------------------------------
// Kernel 1: fused residual add + LayerNorm over C=1024, emit bf16 activations
// ---------------------------------------------------------------------------
__global__ __launch_bounds__(256) void fused_add_ln_bf16(
    const float* __restrict__ x1, const float* __restrict__ x2,
    const float* __restrict__ gamma, const float* __restrict__ beta,
    unsigned short* __restrict__ out) {
  const int row = blockIdx.x;
  const int t   = threadIdx.x;
  const size_t base = (size_t)row * K_DIM + (size_t)t * 4;

  float4 a = *(const float4*)(x1 + base);
  float4 b = *(const float4*)(x2 + base);
  float v0 = a.x + b.x, v1 = a.y + b.y, v2 = a.z + b.z, v3 = a.w + b.w;

  float s = v0 + v1 + v2 + v3;
  float q = v0 * v0 + v1 * v1 + v2 * v2 + v3 * v3;
  #pragma unroll
  for (int off = 16; off > 0; off >>= 1) {   // wave32 butterfly
    s += __shfl_down(s, off, 32);
    q += __shfl_down(q, off, 32);
  }

  __shared__ float ssum[8], ssq[8];
  __shared__ float smu, srstd;
  const int lane = t & 31, wid = t >> 5;
  if (lane == 0) { ssum[wid] = s; ssq[wid] = q; }
  __syncthreads();
  if (t == 0) {
    float ts = 0.f, tq = 0.f;
    #pragma unroll
    for (int i = 0; i < 8; ++i) { ts += ssum[i]; tq += ssq[i]; }
    float mu  = ts * (1.0f / K_DIM);
    float var = tq * (1.0f / K_DIM) - mu * mu;
    smu   = mu;
    srstd = rsqrtf(var + 1e-5f);
  }
  __syncthreads();
  const float mu = smu, rstd = srstd;

  const int c = t * 4;
  float4 g = *(const float4*)(gamma + c);
  float4 e = *(const float4*)(beta + c);
  ushort4 o;
  o.x = f32_to_bf16((v0 - mu) * rstd * g.x + e.x);
  o.y = f32_to_bf16((v1 - mu) * rstd * g.y + e.y);
  o.z = f32_to_bf16((v2 - mu) * rstd * g.z + e.z);
  o.w = f32_to_bf16((v3 - mu) * rstd * g.w + e.w);
  *(ushort4*)(out + base) = o;
}

// ---------------------------------------------------------------------------
// Kernel 2: W [F,K] fp32 -> bf16 (K-contiguous, same layout)
// ---------------------------------------------------------------------------
__global__ __launch_bounds__(256) void w_to_bf16(
    const float* __restrict__ w, unsigned short* __restrict__ o) {
  const size_t i = ((size_t)blockIdx.x * 256 + threadIdx.x) * 4;
  float4 v = *(const float4*)(w + i);
  ushort4 u;
  u.x = f32_to_bf16(v.x); u.y = f32_to_bf16(v.y);
  u.z = f32_to_bf16(v.z); u.w = f32_to_bf16(v.w);
  *(ushort4*)(o + i) = u;
}

// ---------------------------------------------------------------------------
// Kernel 3: bf16 WMMA GEMM (out[m,n] = sum_k A[m,k]*B[n,k]) + bias + erf-GELU
// 256 threads = 8 waves in a 2(M) x 4(N) grid; each wave computes 32x64 =
// 2x4 v_wmma_f32_16x16x32_bf16 tiles. Double-buffered LDS; async global->LDS
// when the toolchain exposes the CDNA5 builtins.
// ---------------------------------------------------------------------------
__global__ __launch_bounds__(256) void gemm_bias_gelu(
    const unsigned short* __restrict__ A, const unsigned short* __restrict__ B,
    const float* __restrict__ bias, float* __restrict__ out) {
  __shared__ unsigned short As[2][BM * LDA];   // 10 KB
  __shared__ unsigned short Bs[2][BN * LDB];   // 40 KB

  const int t    = threadIdx.x;
  const int lane = t & 31;
  const int wid  = t >> 5;
  const int waveM = wid >> 2;   // 0..1 -> 32-row band
  const int waveN = wid & 3;    // 0..3 -> 64-col band
  const int l16  = lane & 15;
  const int lhi  = lane >> 4;   // half-wave select

  const int m0 = blockIdx.y * BM;
  const int n0 = blockIdx.x * BN;

  // global->LDS staging (all 16B chunks, fully coalesced)
  const int arow = t >> 2;            // 0..63
  const int acol = (t & 3) * 8;       // 0,8,16,24 (bf16 elems)
  const int brow = t;                 // 0..255 : one full 32-elem row / thread

  const unsigned short* aptr = A + (size_t)(m0 + arow) * K_DIM + acol;
  const unsigned short* bptr = B + (size_t)(n0 + brow) * K_DIM;

  unsigned short* aLds0 = &As[0][arow * LDA + acol];
  unsigned short* bLds0 = &Bs[0][brow * LDB];
  unsigned short* aLds1 = &As[1][arow * LDA + acol];
  unsigned short* bLds1 = &Bs[1][brow * LDB];

  // ---- prologue: stage k-tile 0 into buffer 0 ----
#ifdef USE_ASYNC_LDS
  {
    async_cp16(aptr, aLds0);
    #pragma unroll
    for (int c = 0; c < 4; ++c) async_cp16(bptr + c * 8, bLds0 + c * 8);
    __builtin_amdgcn_s_wait_asynccnt(0);
  }
#else
  {
    uint4 av = *(const uint4*)(aptr);
    *(uint4*)aLds0 = av;
    #pragma unroll
    for (int c = 0; c < 4; ++c)
      *(uint4*)(bLds0 + c * 8) = *(const uint4*)(bptr + c * 8);
  }
#endif
  __syncthreads();

  v8f acc[2][4] = {};
  // ISA 16-bit fragment layouts (wave32):
  //  A 16x32: lanes 0-15 hold K=0..7 & 16..23, lanes 16-31 hold K=8..15 & 24..31
  //  B 32x16: lanes 0-15 (N=lane) hold K=0..7 & 16..23, lanes 16-31 hold K=8..15 & 24..31
  const int aks = lhi ? 8 : 0;
  const int bks = lhi ? 8 : 0;

  int buf = 0;
  #pragma unroll 1
  for (int kk = 0; kk < KSTEPS; ++kk) {
    const bool more = (kk + 1) < KSTEPS;
    unsigned short* aDst = buf ? aLds0 : aLds1;  // stage into the other buffer
    unsigned short* bDst = buf ? bLds0 : bLds1;

#ifdef USE_ASYNC_LDS
    if (more) {   // async writes go straight to LDS; no VGPR staging
      const int kn = (kk + 1) * BK;
      async_cp16(aptr + kn, aDst);
      #pragma unroll
      for (int c = 0; c < 4; ++c) async_cp16(bptr + kn + c * 8, bDst + c * 8);
    }
#else
    uint4 av, bv[4];
    if (more) {
      const int kn = (kk + 1) * BK;
      av = *(const uint4*)(aptr + kn);
      #pragma unroll
      for (int c = 0; c < 4; ++c) bv[c] = *(const uint4*)(bptr + kn + c * 8);
    }
#endif
    if (kk + 2 < KSTEPS) {   // lowers to global_prefetch_b8 (L2 warm)
      __builtin_prefetch(aptr + (kk + 2) * BK, 0, 1);
      __builtin_prefetch(bptr + (kk + 2) * BK, 0, 1);
    }

    // ---- fragment loads + 8 WMMAs on current buffer ----
    union Frag { bfrag_t v; uint4 u[2]; };
    Frag af[2], bfv[4];
    #pragma unroll
    for (int mi = 0; mi < 2; ++mi) {
      const unsigned short* p = &As[buf][(waveM * 32 + mi * 16 + l16) * LDA + aks];
      af[mi].u[0] = *(const uint4*)(p);        // K=aks..aks+7
      af[mi].u[1] = *(const uint4*)(p + 16);   // K=aks+16..aks+23
    }
    #pragma unroll
    for (int ni = 0; ni < 4; ++ni) {
      const unsigned short* p = &Bs[buf][(waveN * 64 + ni * 16 + l16) * LDB + bks];
      bfv[ni].u[0] = *(const uint4*)(p);       // K=bks..bks+7
      bfv[ni].u[1] = *(const uint4*)(p + 16);  // K=bks+16..bks+23
    }

    #pragma unroll
    for (int mi = 0; mi < 2; ++mi)
      #pragma unroll
      for (int ni = 0; ni < 4; ++ni)
        acc[mi][ni] = __builtin_amdgcn_wmma_f32_16x16x32_bf16(
            false, af[mi].v, false, bfv[ni].v,
            (short)0, acc[mi][ni], false, false);

#ifdef USE_ASYNC_LDS
    if (more) __builtin_amdgcn_s_wait_asynccnt(0);
#else
    if (more) {
      *(uint4*)aDst = av;
      #pragma unroll
      for (int c = 0; c < 4; ++c) *(uint4*)(bDst + c * 8) = bv[c];
    }
#endif
    __syncthreads();
    buf ^= 1;
  }

  // Epilogue: D layout = 8 VGPRs; VGPR r: lanes 0-15 -> M=r, lanes 16-31 -> M=8+r
  #pragma unroll
  for (int mi = 0; mi < 2; ++mi) {
    const int mbase = m0 + waveM * 32 + mi * 16 + lhi * 8;
    #pragma unroll
    for (int ni = 0; ni < 4; ++ni) {
      const int n  = n0 + waveN * 64 + ni * 16 + l16;
      const float bn = bias[n];
      #pragma unroll
      for (int r = 0; r < 8; ++r) {
        float y = acc[mi][ni][r] + bn;
        out[(size_t)(mbase + r) * N_DIM + n] = gelu_erf(y);
      }
    }
  }
}

// ---------------------------------------------------------------------------
extern "C" void kernel_launch(void* const* d_in, const int* in_sizes, int n_in,
                              void* d_out, int out_size, void* d_ws, size_t ws_size,
                              hipStream_t stream) {
  const float* x556  = (const float*)d_in[0];
  const float* x570  = (const float*)d_in[1];
  const float* gamma = (const float*)d_in[2];
  const float* beta  = (const float*)d_in[3];
  const float* W     = (const float*)d_in[4];
  const float* bias  = (const float*)d_in[5];
  float* out = (float*)d_out;

  unsigned short* Abf = (unsigned short*)d_ws;                                       // 25.7 MB
  unsigned short* Bbf = (unsigned short*)((char*)d_ws + (size_t)M_ROWS * K_DIM * 2); // 8.4 MB

  fused_add_ln_bf16<<<M_ROWS, 256, 0, stream>>>(x556, x570, gamma, beta, Abf);
  w_to_bf16<<<(N_DIM * K_DIM) / 1024, 256, 0, stream>>>(W, Bbf);
  gemm_bias_gelu<<<dim3(N_DIM / BN, M_ROWS / BM), 256, 0, stream>>>(Abf, Bbf, bias, out);
}
